// HiddenGATEncoder_3513283248865
// MI455X (gfx1250) — matrix operand order repaired and checked by reading, not compiled, not run on hardware
//
#include <hip/hip_runtime.h>
#include <hip/hip_bf16.h>

// ---------------- problem constants ----------------
constexpr int N_NODES = 50000;
constexpr int N_EDGES = 800000;
constexpr int ET      = N_EDGES + N_NODES;   // edges + self loops = 850000
constexpr int HEADS   = 3;
constexpr int C       = 64;
constexpr int F       = HEADS * C;           // 192
constexpr int F_IN    = 256;

typedef __attribute__((ext_vector_type(16))) _Float16 v16h;
typedef __attribute__((ext_vector_type(8)))  float    v8f;

// pack 8 consecutive f32 (two float4) into v16h elements [base, base+8)
__device__ inline void pack8(v16h& v, int base, float4 p, float4 q) {
  v[base + 0] = (_Float16)p.x; v[base + 1] = (_Float16)p.y;
  v[base + 2] = (_Float16)p.z; v[base + 3] = (_Float16)p.w;
  v[base + 4] = (_Float16)q.x; v[base + 5] = (_Float16)q.y;
  v[base + 6] = (_Float16)q.z; v[base + 7] = (_Float16)q.w;
}

// ---------------- WMMA GEMM: OUT[M,Nout] = X[M,K] @ W[Nout,K]^T ----------------
// Each wave32 computes NT adjacent 16x16 output tiles in the N direction,
// reusing one A fragment for NT WMMAs. K is compile-time -> full unroll.
// Fragment layouts per CDNA5 ISA 7.12.2:
//   A(16x32 f16): lane l, elem j -> (m=l&15, k=16*(j>>3)+8*(l>>4)+(j&7))
//   B(32x16 f16): lane l, elem j -> (k=16*(j>>3)+8*(l>>4)+(j&7), n=l&15)
//   D(16x16 f32): lane l, vgpr j -> (m=8*(l>>4)+j, n=l&15)
template <int K, int NT>
__global__ __launch_bounds__(128) void wmma_gemm_xwT(
    const float* __restrict__ X, const float* __restrict__ W,
    float* __restrict__ OUT, int M, int Nout) {
  int wave = (int)((blockIdx.x * blockDim.x + threadIdx.x) >> 5);
  int lane = threadIdx.x & 31;
  int groupsN = Nout / (16 * NT);
  int mt = wave / groupsN;
  int ng = wave - mt * groupsN;
  if (mt * 16 >= M) return;                 // wave-uniform exit (EXEC stays all-ones)

  int l15  = lane & 15;
  int half = lane >> 4;

  const float* xrow  = X + (size_t)(mt * 16 + l15) * K + 8 * half;
  const float* wbase = W + (size_t)(ng * 16 * NT + l15) * K + 8 * half;

  v8f acc[NT];
#pragma unroll
  for (int t = 0; t < NT; ++t) acc[t] = (v8f){};

#pragma unroll
  for (int kb = 0; kb < K; kb += 32) {
    // A fragment: two contiguous 8-float runs -> 4x b128 loads
    v16h a;
    pack8(a, 0, *(const float4*)(xrow + kb),      *(const float4*)(xrow + kb + 4));
    pack8(a, 8, *(const float4*)(xrow + kb + 16), *(const float4*)(xrow + kb + 20));
#pragma unroll
    for (int t = 0; t < NT; ++t) {
      const float* wrow = wbase + (size_t)t * 16 * K;
      v16h b;   // two contiguous 8-float runs -> 4x b128 loads
      pack8(b, 0, *(const float4*)(wrow + kb),     *(const float4*)(wrow + kb + 4));
      pack8(b, 8, *(const float4*)(wrow + kb + 16), *(const float4*)(wrow + kb + 20));
      acc[t] = __builtin_amdgcn_wmma_f32_16x16x32_f16(
          /*neg_a=*/false, a, /*neg_b=*/false, b,
          /*c_mod=*/(short)0, acc[t], /*reuse_a=*/false, /*reuse_b=*/false);
    }
  }

#pragma unroll
  for (int t = 0; t < NT; ++t) {
    float* orow = OUT + (size_t)(mt * 16 + 8 * half) * Nout + (ng * NT + t) * 16 + l15;
#pragma unroll
    for (int j = 0; j < 8; ++j) orow[(size_t)j * Nout] = acc[t][j];
  }
}

// ---------------- helpers ----------------
__device__ inline void edge_sd(const int* __restrict__ ei, int e, int& s, int& d) {
  if (e < N_EDGES) { s = ei[e]; d = ei[N_EDGES + e]; }
  else             { s = e - N_EDGES; d = s; }           // self loops
}

__device__ inline void atomicMaxF(float* addr, float val) {
  unsigned* ua = (unsigned*)addr;
  unsigned old = *ua;
  while (true) {
    float cur = __uint_as_float(old);
    if (cur >= val) break;
    unsigned assumed = old;
    old = atomicCAS(ua, assumed, __float_as_uint(val));
    if (old == assumed) break;
  }
}

// ---------------- attention coefficients: al_src/al_dst [N,H] ----------------
__global__ void attn_coef(const float* __restrict__ hlin,
                          const float* __restrict__ aS, const float* __restrict__ aD,
                          float* __restrict__ alS, float* __restrict__ alD, int total) {
  int i = blockIdx.x * blockDim.x + threadIdx.x;     // over N*H
  if (i >= total) return;
  int node = i / HEADS, head = i - node * HEADS;
  const float* hrow = hlin + (size_t)node * F + head * C;
  const float* as = aS + head * C;
  const float* ad = aD + head * C;
  float s0 = 0.f, s1 = 0.f;
#pragma unroll 8
  for (int c = 0; c < C; ++c) { float hv = hrow[c]; s0 += hv * as[c]; s1 += hv * ad[c]; }
  alS[i] = s0; alD[i] = s1;
}

// ---------------- per-layer buffer init ----------------
__global__ void init_bufs(float* __restrict__ agg, float* __restrict__ emax,
                          float* __restrict__ den, int nF, int nH) {
  int i = blockIdx.x * blockDim.x + threadIdx.x;
  if (i < nF) agg[i] = 0.f;
  if (i < nH) { emax[i] = -3.402823466e38f; den[i] = 0.f; }
}

// ---------------- edge pass 1: leaky-relu logits + segment max ----------------
__global__ void edge_logit_max(const int* __restrict__ ei,
                               const float* __restrict__ alS, const float* __restrict__ alD,
                               float* __restrict__ eb, float* __restrict__ emax, int total) {
  int i = blockIdx.x * blockDim.x + threadIdx.x;     // over ET*H
  if (i >= total) return;
  int e = i / HEADS, head = i - e * HEADS;
  int s, d; edge_sd(ei, e, s, d);
  float v = alS[s * HEADS + head] + alD[d * HEADS + head];
  v = v > 0.f ? v : 0.2f * v;                        // leaky_relu, slope 0.2
  eb[i] = v;
  atomicMaxF(&emax[d * HEADS + head], v);
}

// ---------------- edge pass 2: exp + segment sum ----------------
__global__ void edge_exp_sum(const int* __restrict__ ei, float* __restrict__ eb,
                             const float* __restrict__ emax, float* __restrict__ den, int total) {
  int i = blockIdx.x * blockDim.x + threadIdx.x;     // over ET*H
  if (i >= total) return;
  int e = i / HEADS, head = i - e * HEADS;
  int s, d; edge_sd(ei, e, s, d);
  float ex = __expf(eb[i] - emax[d * HEADS + head]);
  eb[i] = ex;
  atomicAdd(&den[d * HEADS + head], ex);
}

// ---------------- edge pass 3: weighted scatter-add of messages ----------------
__global__ void edge_aggregate(const int* __restrict__ ei, const float* __restrict__ eb,
                               const float* __restrict__ den, const float* __restrict__ hlin,
                               float* __restrict__ agg, int total) {
  int i = blockIdx.x * blockDim.x + threadIdx.x;     // over ET*H*C (163.2M)
  if (i >= total) return;
  int c    = i & (C - 1);
  int rest = i >> 6;                                 // = e*HEADS + head
  int e    = rest / HEADS, head = rest - e * HEADS;
  int s, d; edge_sd(ei, e, s, d);
  float alpha = eb[rest] / (den[d * HEADS + head] + 1e-16f);
  float val   = hlin[(size_t)s * F + head * C + c] * alpha;
  atomicAdd(&agg[(size_t)d * F + head * C + c], val);
}

// ---------------- finalize layer 1: +bias, ReLU (concat=True) ----------------
__global__ void finalize_relu_bias(const float* __restrict__ agg, const float* __restrict__ b,
                                   float* __restrict__ h1, int total) {
  int i = blockIdx.x * blockDim.x + threadIdx.x;     // over N*F
  if (i >= total) return;
  int col = i % F;
  float v = agg[i] + b[col];
  h1[i] = v > 0.f ? v : 0.f;
}

// ---------------- finalize mu/logvar: mean over heads + bias ----------------
__global__ void finalize_mean(const float* __restrict__ agg, const float* __restrict__ b,
                              float* __restrict__ out, int total) {
  int i = blockIdx.x * blockDim.x + threadIdx.x;     // over N*C
  if (i >= total) return;
  int c = i & (C - 1);
  int node = i >> 6;
  const float* r = agg + (size_t)node * F;
  out[i] = (r[c] + r[C + c] + r[2 * C + c]) * (1.0f / 3.0f) + b[c];
}

// ---------------- host orchestration ----------------
static inline int cdiv(long long a, int b) { return (int)((a + b - 1) / b); }

extern "C" void kernel_launch(void* const* d_in, const int* in_sizes, int n_in,
                              void* d_out, int out_size, void* d_ws, size_t ws_size,
                              hipStream_t stream) {
  (void)in_sizes; (void)n_in; (void)out_size; (void)ws_size;
  const float* x    = (const float*)d_in[0];
  const int*   ei   = (const int*)d_in[1];
  const float* W1   = (const float*)d_in[2];
  const float* a1s  = (const float*)d_in[3];
  const float* a1d  = (const float*)d_in[4];
  const float* b1   = (const float*)d_in[5];
  const float* Wmu  = (const float*)d_in[6];
  const float* amus = (const float*)d_in[7];
  const float* amud = (const float*)d_in[8];
  const float* bmu  = (const float*)d_in[9];
  const float* Wlv  = (const float*)d_in[10];
  const float* alvs = (const float*)d_in[11];
  const float* alvd = (const float*)d_in[12];
  const float* blv  = (const float*)d_in[13];
  float* out = (float*)d_out;

  // workspace carve-up (256B aligned)
  char* ws = (char*)d_ws;
  size_t off = 0;
  auto alloc = [&](size_t nfl) {
    float* p = (float*)(ws + off);
    off += ((nfl * sizeof(float) + 255) / 256) * 256;
    return p;
  };
  float* hlin = alloc((size_t)N_NODES * F);   // linear transform of current layer
  float* agg  = alloc((size_t)N_NODES * F);   // aggregation target
  float* h1   = alloc((size_t)N_NODES * F);   // layer-1 output (persistent)
  float* eb   = alloc((size_t)ET * HEADS);    // edge logits / exp
  float* emax = alloc((size_t)N_NODES * HEADS);
  float* den  = alloc((size_t)N_NODES * HEADS);
  float* alS  = alloc((size_t)N_NODES * HEADS);
  float* alD  = alloc((size_t)N_NODES * HEADS);

  const int nH   = N_NODES * HEADS;           // 150,000
  const int nF   = N_NODES * F;               // 9,600,000
  const int etH  = ET * HEADS;                // 2,550,000
  const int etHC = ET * HEADS * C;            // 163,200,000
  const int nC   = N_NODES * C;               // 3,200,000

  constexpr int NT = 4;                       // n-tiles per wave (Nout=192=3*4*16)
  const int tiles = (N_NODES / 16) * (F / (16 * NT));   // 3125 * 3 = 9375 waves

  auto run_edges = [&](const float* hl) {
    init_bufs<<<cdiv(nF, 256), 256, 0, stream>>>(agg, emax, den, nF, nH);
    edge_logit_max<<<cdiv(etH, 256), 256, 0, stream>>>(ei, alS, alD, eb, emax, etH);
    edge_exp_sum<<<cdiv(etH, 256), 256, 0, stream>>>(ei, eb, emax, den, etH);
    edge_aggregate<<<cdiv(etHC, 256), 256, 0, stream>>>(ei, eb, den, hl, agg, etHC);
  };

  // ---- layer 1 (concat=True, then ReLU) ----
  wmma_gemm_xwT<F_IN, NT><<<cdiv(tiles, 4), 128, 0, stream>>>(x, W1, hlin, N_NODES, F);
  attn_coef<<<cdiv(nH, 256), 256, 0, stream>>>(hlin, a1s, a1d, alS, alD, nH);
  run_edges(hlin);
  finalize_relu_bias<<<cdiv(nF, 256), 256, 0, stream>>>(agg, b1, h1, nF);

  // ---- mu layer (concat=False: mean over heads) ----
  wmma_gemm_xwT<F, NT><<<cdiv(tiles, 4), 128, 0, stream>>>(h1, Wmu, hlin, N_NODES, F);
  attn_coef<<<cdiv(nH, 256), 256, 0, stream>>>(hlin, amus, amud, alS, alD, nH);
  run_edges(hlin);
  finalize_mean<<<cdiv(nC, 256), 256, 0, stream>>>(agg, bmu, out, nC);

  // ---- logvar layer ----
  wmma_gemm_xwT<F, NT><<<cdiv(tiles, 4), 128, 0, stream>>>(h1, Wlv, hlin, N_NODES, F);
  attn_coef<<<cdiv(nH, 256), 256, 0, stream>>>(hlin, alvs, alvd, alS, alD, nH);
  run_edges(hlin);
  finalize_mean<<<cdiv(nC, 256), 256, 0, stream>>>(agg, blv, out + (size_t)nC, nC);
}
